// MHA_11888469475605
// MI455X (gfx1250) — matrix operand order repaired and checked by reading, not compiled, not run on hardware
//
#include <hip/hip_runtime.h>
#include <hip/hip_bf16.h>

typedef _Float16 f16;
typedef __attribute__((ext_vector_type(8)))  _Float16 v8h;
typedef __attribute__((ext_vector_type(16))) _Float16 v16h;
typedef __attribute__((ext_vector_type(8)))  float    v8f;

// Problem constants (fixed by reference setup_inputs)
constexpr int Bb  = 4;
constexpr int Nn  = 2048;
constexpr int Cc  = 1024;
constexpr int Hh  = 16;
constexpr int HD  = 64;
constexpr int C3  = 3 * Cc;           // 3072
constexpr int Mrows = Bb * Nn;        // 8192

#if __has_builtin(__builtin_amdgcn_global_load_async_to_lds_b128)
#define USE_ASYNC_LDS 1
#else
#define USE_ASYNC_LDS 0
#endif

// The async-to-LDS builtin takes pointers to 16-byte int vectors:
//   param0: global (AS1) v4i*   param1: LDS (AS3) v4i*   then imm offset, imm cpol
typedef __attribute__((vector_size(16))) int v4i_t;
typedef __attribute__((address_space(1))) v4i_t* gv4p;
typedef __attribute__((address_space(3))) v4i_t* sv4p;

// 16-byte global -> LDS copy: async DMA (ASYNCcnt) on gfx1250 toolchains that
// declare the builtin, else synchronous VGPR round-trip.
__device__ inline void cp16_g2l(const f16* g, f16* l) {
#if USE_ASYNC_LDS
  __builtin_amdgcn_global_load_async_to_lds_b128((gv4p)g, (sv4p)l, 0, 0);
#else
  *(uint4*)l = *(const uint4*)g;
#endif
}

__device__ inline void wait_async0() {
#if USE_ASYNC_LDS
#if __has_builtin(__builtin_amdgcn_s_wait_asynccnt)
  __builtin_amdgcn_s_wait_asynccnt(0);
#else
  asm volatile("s_wait_asynccnt 0x0" ::: "memory");
#endif
#endif
}

__device__ inline v8f vzero8f() {
  v8f z = {0.f, 0.f, 0.f, 0.f, 0.f, 0.f, 0.f, 0.f};
  return z;
}

// Assemble a v16h WMMA fragment from two 16-byte-aligned pieces.
__device__ inline v16h ldfrag(const f16* p0, const f16* p1) {
  v8h lo = *(const v8h*)p0;
  v8h hi = *(const v8h*)p1;
  return __builtin_shufflevector(lo, hi, 0, 1, 2, 3, 4, 5, 6, 7,
                                 8, 9, 10, 11, 12, 13, 14, 15);
}

__device__ inline v8f wmma_f16(v16h a, v16h b, v8f c) {
  return __builtin_amdgcn_wmma_f32_16x16x32_f16(
      /*neg_a=*/false, a, /*neg_b=*/false, b,
      /*c_mod=*/(short)0, c, /*reuse_a=*/false, /*reuse_b=*/false);
}

// ---------------------------------------------------------------------------
// Kernel 1: x = inputs + sinusoidal PE, converted to f16.
// ---------------------------------------------------------------------------
__global__ __launch_bounds__(256) void k_pe_convert(const float* __restrict__ x,
                                                    f16* __restrict__ xh) {
  int i = blockIdx.x * 256 + threadIdx.x;           // over Mrows*Cc
  int c = i & (Cc - 1);
  int n = (i >> 10) & (Nn - 1);                     // Cc == 1024 == 2^10
  int i2 = c & ~1;
  // rate = 10000^{-i2/C} = exp(-i2 * ln(10000)/C)
  float rate = __expf(-(float)i2 * (9.210340371976184f / (float)Cc));
  float ang  = (float)n * rate;
  float pe   = (c & 1) ? __cosf(ang) : __sinf(ang);
  xh[i] = (f16)(x[i] + pe);
}

// ---------------------------------------------------------------------------
// Kernel 2: wh[nc][k] = (f16) W_qkv[k][nc]   (transposed, f16)
// ---------------------------------------------------------------------------
__global__ __launch_bounds__(256) void k_w_convert(const float* __restrict__ W,
                                                   f16* __restrict__ wh) {
  int i  = blockIdx.x * 256 + threadIdx.x;          // over C3*Cc
  int nc = i >> 10;                                 // 0..3071
  int kk = i & (Cc - 1);
  wh[i] = (f16)W[kk * C3 + nc];
}

// ---------------------------------------------------------------------------
// Kernel 3: QKV GEMM. xh[M,C] * W[C,3C] -> scatter into Q,K (row-major per
// head) and Vt (hd-major per head). 256 threads = 8 waves (4 x 2),
// 128x128 output tile per block, K-chunk = 32, double-buffered async LDS.
// ---------------------------------------------------------------------------
__global__ __launch_bounds__(256) void k_qkv_gemm(const f16* __restrict__ xh,
                                                  const f16* __restrict__ wh,
                                                  f16* __restrict__ Qo,
                                                  f16* __restrict__ Ko,
                                                  f16* __restrict__ Vto) {
  __shared__ alignas(16) f16 As[2][128 * 32];   // [row][k]   2 x 8 KB
  __shared__ alignas(16) f16 Bs[2][128 * 32];   // [col][k]   2 x 8 KB

  const int m0 = blockIdx.x * 128;
  const int n0 = blockIdx.y * 128;
  const int tid  = threadIdx.x;
  const int lane = tid & 31;
  const int wave = tid >> 5;
  const int wm = wave & 3;        // 0..3 : 32-row strip
  const int wn = wave >> 2;       // 0..1 : 64-col strip
  const int hi16 = lane >> 4;
  const int lm   = lane & 15;

  // Issue the (async) tile copies for K-chunk k0 into buffer `bf`.
  auto issue_tile = [&](int bf, int k0) {
#pragma unroll
    for (int gg = 0; gg < 2; ++gg) {
      int g    = tid + gg * 256;       // 0..511
      int row  = g >> 2;
      int col8 = (g & 3) * 8;
      cp16_g2l(&xh[(size_t)(m0 + row) * Cc + k0 + col8],
               &As[bf][row * 32 + col8]);
      cp16_g2l(&wh[(size_t)(n0 + row) * Cc + k0 + col8],
               &Bs[bf][row * 32 + col8]);
    }
  };

  v8f acc[2][4];
#pragma unroll
  for (int mt = 0; mt < 2; ++mt)
#pragma unroll
    for (int nt = 0; nt < 4; ++nt) acc[mt][nt] = vzero8f();

  issue_tile(0, 0);
  int bf = 0;
  for (int k0 = 0; k0 < Cc; k0 += 32) {
    wait_async0();
    __syncthreads();                    // buffer `bf` ready; prior reads done
    if (k0 + 32 < Cc) issue_tile(bf ^ 1, k0 + 32);

    v16h afrag[2], bfrag[4];
#pragma unroll
    for (int mt = 0; mt < 2; ++mt) {
      int row = wm * 32 + mt * 16 + lm;
      afrag[mt] = ldfrag(&As[bf][row * 32 + hi16 * 8],
                         &As[bf][row * 32 + 16 + hi16 * 8]);
    }
#pragma unroll
    for (int nt = 0; nt < 4; ++nt) {
      int row = wn * 64 + nt * 16 + lm;
      bfrag[nt] = ldfrag(&Bs[bf][row * 32 + hi16 * 8],
                         &Bs[bf][row * 32 + 16 + hi16 * 8]);
    }
#pragma unroll
    for (int mt = 0; mt < 2; ++mt)
#pragma unroll
      for (int nt = 0; nt < 4; ++nt)
        acc[mt][nt] = wmma_f16(afrag[mt], bfrag[nt], acc[mt][nt]);
    bf ^= 1;
  }

  // Epilogue: scatter f16 into Q / K / Vt per-head layouts.
#pragma unroll
  for (int mt = 0; mt < 2; ++mt) {
#pragma unroll
    for (int nt = 0; nt < 4; ++nt) {
#pragma unroll
      for (int r = 0; r < 8; ++r) {
        int gm = m0 + wm * 32 + mt * 16 + r + 8 * hi16;   // 0..8191
        int gc = n0 + wn * 64 + nt * 16 + lm;             // 0..3071
        int bi = gm >> 11;            // /2048
        int ni = gm & (Nn - 1);
        int s  = gc >> 10;            // 0=q 1=k 2=v
        int rem = gc & (Cc - 1);
        int h  = rem >> 6;
        int e  = rem & 63;
        f16 val = (f16)acc[mt][nt][r];
        if (s == 0) {
          Qo[(((size_t)(bi * Hh + h) * Nn + ni) << 6) + e] = val;
        } else if (s == 1) {
          Ko[(((size_t)(bi * Hh + h) * Nn + ni) << 6) + e] = val;
        } else {
          Vto[((size_t)(bi * Hh + h) * HD + e) * Nn + ni] = val;
        }
      }
    }
  }
}

// ---------------------------------------------------------------------------
// Kernel 4: flash attention. Block = 128 threads = 4 waves; block handles
// (b, h, 64 query rows); each wave owns 16 query rows. Key tiles of 128,
// double-buffered async LDS staging for K / V^T tiles.
// ---------------------------------------------------------------------------
__global__ __launch_bounds__(128) void k_attn(const f16* __restrict__ Qg,
                                              const f16* __restrict__ Kg,
                                              const f16* __restrict__ Vtg,
                                              const float* __restrict__ mask,
                                              float* __restrict__ out) {
  __shared__ alignas(16) f16 Ks[2][128 * 64];    // [key][hd]   2 x 16 KB
  __shared__ alignas(16) f16 Vs[2][64 * 128];    // [hd][key]   2 x 16 KB
  __shared__ alignas(16) f16 Ps[4][16 * 128];    // per-wave P  16 KB
  __shared__ float mk[2][128];

  const int b = blockIdx.z;
  const int h = blockIdx.y;
  const int q0 = blockIdx.x * 64;
  const int tid  = threadIdx.x;
  const int lane = tid & 31;
  const int wave = tid >> 5;
  const int hi16 = lane >> 4;
  const int lm   = lane & 15;
  const float scale = 0.125f;    // 64^-0.5

  const f16* Qh  = Qg  + (size_t)(b * Hh + h) * Nn * HD;
  const f16* Kh  = Kg  + (size_t)(b * Hh + h) * Nn * HD;
  const f16* Vth = Vtg + (size_t)(b * Hh + h) * HD * Nn;

  // Issue (async) staging of key-tile j into buffer `bf`.
  auto issue_tile = [&](int bf, int j) {
    int t = tid;
#pragma unroll
    for (int u = 0; u < 8; ++u)
      cp16_g2l(&Kh[(size_t)(j * 128 + t) * HD + u * 8],
               &Ks[bf][t * 64 + u * 8]);
    int vrow = t >> 1;
    int voff = (t & 1) * 64;
#pragma unroll
    for (int u = 0; u < 8; ++u)
      cp16_g2l(&Vth[(size_t)vrow * Nn + j * 128 + voff + u * 8],
               &Vs[bf][vrow * 128 + voff + u * 8]);
    mk[bf][t] = mask[b * Nn + j * 128 + t];
  };

  // Q fragments (A-matrix, 16x32 per k-chunk): rows q0+wave*16+lm.
  const int qrow = q0 + wave * 16 + lm;
  v16h qa[2];
#pragma unroll
  for (int kc = 0; kc < 2; ++kc)
    qa[kc] = ldfrag(&Qh[(size_t)qrow * HD + kc * 32 + hi16 * 8],
                    &Qh[(size_t)qrow * HD + kc * 32 + 16 + hi16 * 8]);

  // Query-row mask values for this lane's 8 rows.
  float mq[8];
#pragma unroll
  for (int r = 0; r < 8; ++r)
    mq[r] = mask[b * Nn + q0 + wave * 16 + 8 * hi16 + r];

  float m_i[8], l_i[8];
  v8f o[4];
#pragma unroll
  for (int r = 0; r < 8; ++r) { m_i[r] = -3.0e38f; l_i[r] = 0.f; }
#pragma unroll
  for (int ot = 0; ot < 4; ++ot) o[ot] = vzero8f();

  issue_tile(0, 0);
  int bf = 0;
  for (int j = 0; j < Nn / 128; ++j) {
    wait_async0();
    __syncthreads();                  // buffer `bf` ready; prior reads done
    if (j + 1 < Nn / 128) issue_tile(bf ^ 1, j + 1);

    // --- S = Q K^T for 8 key sub-tiles of 16 ---
    v8f s[8];
#pragma unroll
    for (int jt = 0; jt < 8; ++jt) {
      int brow = jt * 16 + lm;
      v16h kb0 = ldfrag(&Ks[bf][brow * 64 + 0 * 32 + hi16 * 8],
                        &Ks[bf][brow * 64 + 0 * 32 + 16 + hi16 * 8]);
      v16h kb1 = ldfrag(&Ks[bf][brow * 64 + 1 * 32 + hi16 * 8],
                        &Ks[bf][brow * 64 + 1 * 32 + 16 + hi16 * 8]);
      s[jt] = vzero8f();
      s[jt] = wmma_f16(qa[0], kb0, s[jt]);
      s[jt] = wmma_f16(qa[1], kb1, s[jt]);
    }

    // --- scale + additive mask: s*scale - mq*mk ---
    float mkc[8];
#pragma unroll
    for (int jt = 0; jt < 8; ++jt) mkc[jt] = mk[bf][jt * 16 + lm];
#pragma unroll
    for (int jt = 0; jt < 8; ++jt)
#pragma unroll
      for (int r = 0; r < 8; ++r)
        s[jt][r] = s[jt][r] * scale - mq[r] * mkc[jt];

    // --- online softmax: row max over this 128-key tile ---
    float tmax[8];
#pragma unroll
    for (int r = 0; r < 8; ++r) {
      float v = s[0][r];
#pragma unroll
      for (int jt = 1; jt < 8; ++jt) v = fmaxf(v, s[jt][r]);
      v = fmaxf(v, __shfl_xor(v, 1));
      v = fmaxf(v, __shfl_xor(v, 2));
      v = fmaxf(v, __shfl_xor(v, 4));
      v = fmaxf(v, __shfl_xor(v, 8));
      tmax[r] = v;
    }

    float corr[8];
#pragma unroll
    for (int r = 0; r < 8; ++r) {
      float nm = fmaxf(m_i[r], tmax[r]);
      corr[r] = __expf(m_i[r] - nm);
      m_i[r] = nm;
    }

    float tsum[8];
#pragma unroll
    for (int r = 0; r < 8; ++r) tsum[r] = 0.f;
#pragma unroll
    for (int jt = 0; jt < 8; ++jt)
#pragma unroll
      for (int r = 0; r < 8; ++r) {
        float p = __expf(s[jt][r] - m_i[r]);
        s[jt][r] = p;
        tsum[r] += p;
      }
#pragma unroll
    for (int r = 0; r < 8; ++r) {
      float v = tsum[r];
      v += __shfl_xor(v, 1);
      v += __shfl_xor(v, 2);
      v += __shfl_xor(v, 4);
      v += __shfl_xor(v, 8);
      l_i[r] = l_i[r] * corr[r] + v;
    }
#pragma unroll
    for (int ot = 0; ot < 4; ++ot)
#pragma unroll
      for (int r = 0; r < 8; ++r) o[ot][r] *= corr[r];

    // --- re-fragment P via per-wave LDS (C/D layout -> A layout) ---
#pragma unroll
    for (int jt = 0; jt < 8; ++jt)
#pragma unroll
      for (int r = 0; r < 8; ++r)
        Ps[wave][(r + 8 * hi16) * 128 + jt * 16 + lm] = (f16)s[jt][r];

    // --- O += P V : key dim 128 in 4 chunks of 32 ---
#pragma unroll
    for (int kc = 0; kc < 4; ++kc) {
      v16h pf = ldfrag(&Ps[wave][lm * 128 + kc * 32 + hi16 * 8],
                       &Ps[wave][lm * 128 + kc * 32 + 16 + hi16 * 8]);
#pragma unroll
      for (int ot = 0; ot < 4; ++ot) {
        v16h vf = ldfrag(&Vs[bf][(ot * 16 + lm) * 128 + kc * 32 + hi16 * 8],
                         &Vs[bf][(ot * 16 + lm) * 128 + kc * 32 + 16 + hi16 * 8]);
        o[ot] = wmma_f16(pf, vf, o[ot]);
      }
    }
    bf ^= 1;
  }

  // --- normalize and store out[b][row][h*64 + col] (f32) ---
#pragma unroll
  for (int ot = 0; ot < 4; ++ot)
#pragma unroll
    for (int r = 0; r < 8; ++r) {
      int row = q0 + wave * 16 + r + 8 * hi16;
      int col = h * HD + ot * 16 + lm;
      out[((size_t)b * Nn + row) * Cc + col] = o[ot][r] / l_i[r];
    }
}

// ---------------------------------------------------------------------------
extern "C" void kernel_launch(void* const* d_in, const int* in_sizes, int n_in,
                              void* d_out, int out_size, void* d_ws, size_t ws_size,
                              hipStream_t stream) {
  (void)in_sizes; (void)n_in; (void)out_size; (void)ws_size;
  const float* inputs = (const float*)d_in[0];   // [4,2048,1024]
  const float* maskm  = (const float*)d_in[1];   // [4,2048]
  const float* Wqkv   = (const float*)d_in[2];   // [1024,3072]
  float* out = (float*)d_out;                    // [4,2048,1024]

  char* ws = (char*)d_ws;
  const size_t xh_off = 0;
  const size_t wh_off = xh_off + (size_t)Mrows * Cc * 2;     // 16 MB
  const size_t q_off  = wh_off + (size_t)C3 * Cc * 2;        // +6 MB
  const size_t k_off  = q_off  + (size_t)Bb * Hh * Nn * HD * 2;
  const size_t v_off  = k_off  + (size_t)Bb * Hh * Nn * HD * 2;

  f16* xh = (f16*)(ws + xh_off);
  f16* wh = (f16*)(ws + wh_off);
  f16* Qp = (f16*)(ws + q_off);
  f16* Kp = (f16*)(ws + k_off);
  f16* Vp = (f16*)(ws + v_off);

  k_pe_convert<<<(Mrows * Cc) / 256, 256, 0, stream>>>(inputs, xh);
  k_w_convert<<<(C3 * Cc) / 256, 256, 0, stream>>>(Wqkv, wh);
  k_qkv_gemm<<<dim3(Mrows / 128, C3 / 128), 256, 0, stream>>>(xh, wh, Qp, Kp, Vp);
  k_attn<<<dim3(Nn / 64, Hh, Bb), 128, 0, stream>>>(Qp, Kp, Vp, maskm, out);
}
